// SGFormer_12309376271051
// MI455X (gfx1250) — compile-verified, matrix operand order checked by reading, not run on hardware
//
#include <hip/hip_runtime.h>
#include <hip/hip_bf16.h>

typedef __bf16 bf16;
typedef __attribute__((ext_vector_type(16))) __bf16 v16bf;
typedef __attribute__((ext_vector_type(8)))  __bf16 v8bf;
typedef __attribute__((ext_vector_type(8)))  float  v8f;

#define LN_EPS 1e-5f
#define ATB_SPLITS 25
#define LDP 40   // padded LDS row stride (elements): 80B rows, de-conflicts banks

static const int NN  = 50000;   // nodes
static const int DIN = 512;
static const int HC  = 1024;    // H * C

// ---------------------------------------------------------------------------
// Generic bf16 WMMA GEMM: C[M,Ncols] = A[M,K] @ B[K,Ncols] (+bias), fp32 acc.
// 256 threads = 8 waves; tile 128x128; wave computes 64x32 (4x2 WMMA tiles).
// Double-buffered LDS, register-staged pipeline, one barrier per K-step.
// sA: [128][LDP] (m,k). sB: [128][LDP] transposed (n,k).
// ---------------------------------------------------------------------------
__global__ __launch_bounds__(256) void k_gemm_bf16(
    const bf16* __restrict__ A, const bf16* __restrict__ B,
    const float* __restrict__ bias,
    float* __restrict__ Cf, bf16* __restrict__ Cb,
    int M, int Ncols, int K, int lda, int ldb, int ldc)
{
  __shared__ bf16 sA[2][128 * LDP];
  __shared__ bf16 sB[2][128 * LDP];
  const int tid  = threadIdx.x;
  const int lane = tid & 31;
  const int wave = tid >> 5;
  const int wrow = wave & 1;   // 2 row groups of 64
  const int wcol = wave >> 1;  // 4 col groups of 32
  const int rowBase = blockIdx.y * 128;
  const int colBase = blockIdx.x * 128;

  v8f acc[4][2];
#pragma unroll
  for (int i = 0; i < 4; ++i)
#pragma unroll
    for (int j = 0; j < 2; ++j)
#pragma unroll
      for (int e = 0; e < 8; ++e) acc[i][j][e] = 0.0f;

  v8bf ra[2], rb[2];

  auto loadTiles = [&](int kb) {
#pragma unroll
    for (int u = 0; u < 2; ++u) {
      int t = tid + u * 256;
      int r = t >> 2, c8 = (t & 3) << 3;          // A: 128 rows x 4 chunks
      int gr = rowBase + r; if (gr >= M) gr = M - 1;  // clamp; rows never stored
      ra[u] = *(const v8bf*)&A[(size_t)gr * lda + kb + c8];
      int kk = t >> 4, n0 = (t & 15) << 3;        // B: 32 k-rows x 16 chunks
      rb[u] = *(const v8bf*)&B[(size_t)(kb + kk) * ldb + colBase + n0];
    }
  };
  auto stageTiles = [&](int buf) {
#pragma unroll
    for (int u = 0; u < 2; ++u) {
      int t = tid + u * 256;
      int r = t >> 2, c8 = (t & 3) << 3;
      *(v8bf*)&sA[buf][r * LDP + c8] = ra[u];
      int kk = t >> 4, n0 = (t & 15) << 3;
#pragma unroll
      for (int e = 0; e < 8; ++e) sB[buf][(n0 + e) * LDP + kk] = rb[u][e];
    }
  };
  auto compute = [&](int buf) {
    // 16-bit operand layout: lane<16 holds K {0..7,16..23}, lane>=16 {8..15,24..31}
    v16bf bfrag[2];
#pragma unroll
    for (int j = 0; j < 2; ++j) {
      int base = (wcol * 32 + j * 16 + (lane & 15)) * LDP + ((lane >> 4) << 3);
      v8bf lo = *(v8bf*)&sB[buf][base];
      v8bf hi = *(v8bf*)&sB[buf][base + 16];
#pragma unroll
      for (int e = 0; e < 8; ++e) { bfrag[j][e] = lo[e]; bfrag[j][e + 8] = hi[e]; }
    }
#pragma unroll
    for (int i = 0; i < 4; ++i) {
      int base = (wrow * 64 + i * 16 + (lane & 15)) * LDP + ((lane >> 4) << 3);
      v8bf lo = *(v8bf*)&sA[buf][base];
      v8bf hi = *(v8bf*)&sA[buf][base + 16];
      v16bf afrag;
#pragma unroll
      for (int e = 0; e < 8; ++e) { afrag[e] = lo[e]; afrag[e + 8] = hi[e]; }
#pragma unroll
      for (int j = 0; j < 2; ++j) {
        acc[i][j] = __builtin_amdgcn_wmma_f32_16x16x32_bf16(
            false, afrag, false, bfrag[j], (short)0, acc[i][j], false, false);
      }
    }
  };

  // software pipeline: prologue load, then overlap next-tile loads with WMMA
  loadTiles(0);
  stageTiles(0);
  __syncthreads();
  const int nk = K >> 5;
  for (int it = 0; it < nk; ++it) {
    const bool hasNext = (it + 1 < nk);
    if (hasNext) loadTiles((it + 1) << 5);   // global loads in flight during WMMA
    compute(it & 1);
    if (hasNext) stageTiles((it + 1) & 1);
    __syncthreads();
  }

  const int lrow = (lane >> 4) << 3;  // C/D layout: lanes 16-31 hold M+8
  const int lcol = lane & 15;
#pragma unroll
  for (int i = 0; i < 4; ++i) {
#pragma unroll
    for (int j = 0; j < 2; ++j) {
      int col = colBase + wcol * 32 + j * 16 + lcol;
      float bv = bias ? bias[col] : 0.0f;
#pragma unroll
      for (int e = 0; e < 8; ++e) {
        int row = rowBase + wrow * 64 + i * 16 + lrow + e;
        if (row < M) {
          float val = acc[i][j][e] + bv;
          if (Cf) Cf[(size_t)row * ldc + col] = val;
          if (Cb) Cb[(size_t)row * ldc + col] = (bf16)val;
        }
      }
    }
  }
}

// ---------------------------------------------------------------------------
// kvs = k^T v per head: C[m,n] += sum_l A[l, head*512+m] * B[l, head*512+n]
// Split-K over node rows with fp32 atomic accumulation. 512x512 output/head.
// Same double-buffered pipeline; out-of-range rows zero-filled (contraction!).
// ---------------------------------------------------------------------------
__global__ __launch_bounds__(256) void k_gemm_atb(
    const bf16* __restrict__ A, const bf16* __restrict__ B,
    float* __restrict__ C, int rows, int ld)
{
  __shared__ bf16 sA[2][128 * LDP];  // [m][l]
  __shared__ bf16 sB[2][128 * LDP];  // [n][l]
  const int tid = threadIdx.x, lane = tid & 31, wave = tid >> 5;
  const int wrow = wave & 1, wcol = wave >> 1;
  const int head  = blockIdx.z / ATB_SPLITS;
  const int split = blockIdx.z % ATB_SPLITS;
  const int chunk = (rows + ATB_SPLITS - 1) / ATB_SPLITS;
  const int l0 = split * chunk;
  int l1 = l0 + chunk; if (l1 > rows) l1 = rows;
  const int mBase = blockIdx.y * 128, nBase = blockIdx.x * 128;
  const bf16* Ah = A + head * 512;
  const bf16* Bh = B + head * 512;
  float* Ch = C + (size_t)head * 512 * 512;

  v8f acc[4][2];
#pragma unroll
  for (int i = 0; i < 4; ++i)
#pragma unroll
    for (int j = 0; j < 2; ++j)
#pragma unroll
      for (int e = 0; e < 8; ++e) acc[i][j][e] = 0.0f;

  v8bf ra[2], rb[2];

  auto loadTiles = [&](int lb) {
#pragma unroll
    for (int u = 0; u < 2; ++u) {
      int t = tid + u * 256;
      int l = t >> 4, m0 = (t & 15) << 3;         // 32 l-rows x 16 chunks
      int gl = lb + l;
      v8bf za, zb;
#pragma unroll
      for (int e = 0; e < 8; ++e) { za[e] = (bf16)0.0f; zb[e] = (bf16)0.0f; }
      if (gl < l1) {
        za = *(const v8bf*)&Ah[(size_t)gl * ld + mBase + m0];
        zb = *(const v8bf*)&Bh[(size_t)gl * ld + nBase + m0];
      }
      ra[u] = za; rb[u] = zb;
    }
  };
  auto stageTiles = [&](int buf) {
#pragma unroll
    for (int u = 0; u < 2; ++u) {
      int t = tid + u * 256;
      int l = t >> 4, m0 = (t & 15) << 3;
#pragma unroll
      for (int e = 0; e < 8; ++e) {
        sA[buf][(m0 + e) * LDP + l] = ra[u][e];
        sB[buf][(m0 + e) * LDP + l] = rb[u][e];
      }
    }
  };
  auto compute = [&](int buf) {
    v16bf bfrag[2];
#pragma unroll
    for (int j = 0; j < 2; ++j) {
      int base = (wcol * 32 + j * 16 + (lane & 15)) * LDP + ((lane >> 4) << 3);
      v8bf lo = *(v8bf*)&sB[buf][base];
      v8bf hi = *(v8bf*)&sB[buf][base + 16];
#pragma unroll
      for (int e = 0; e < 8; ++e) { bfrag[j][e] = lo[e]; bfrag[j][e + 8] = hi[e]; }
    }
#pragma unroll
    for (int i = 0; i < 4; ++i) {
      int base = (wrow * 64 + i * 16 + (lane & 15)) * LDP + ((lane >> 4) << 3);
      v8bf lo = *(v8bf*)&sA[buf][base];
      v8bf hi = *(v8bf*)&sA[buf][base + 16];
      v16bf afrag;
#pragma unroll
      for (int e = 0; e < 8; ++e) { afrag[e] = lo[e]; afrag[e + 8] = hi[e]; }
#pragma unroll
      for (int j = 0; j < 2; ++j) {
        acc[i][j] = __builtin_amdgcn_wmma_f32_16x16x32_bf16(
            false, afrag, false, bfrag[j], (short)0, acc[i][j], false, false);
      }
    }
  };

  const int nk = (l1 - l0 + 31) >> 5;
  if (nk > 0) {
    loadTiles(l0);
    stageTiles(0);
    __syncthreads();
    for (int it = 0; it < nk; ++it) {
      const bool hasNext = (it + 1 < nk);
      if (hasNext) loadTiles(l0 + ((it + 1) << 5));
      compute(it & 1);
      if (hasNext) stageTiles((it + 1) & 1);
      __syncthreads();
    }
  }

  const int lrow = (lane >> 4) << 3;
  const int lcol = lane & 15;
#pragma unroll
  for (int i = 0; i < 4; ++i)
#pragma unroll
    for (int j = 0; j < 2; ++j)
#pragma unroll
      for (int e = 0; e < 8; ++e) {
        int m = mBase + wrow * 64 + i * 16 + lrow + e;
        int n = nBase + wcol * 32 + j * 16 + lcol;
        atomicAdd(&Ch[(size_t)m * 512 + n], acc[i][j][e]);
      }
}

// ---------------------------------------------------------------------------
// Elementwise / reduction helpers
// ---------------------------------------------------------------------------
__global__ void k_f32_to_bf16(const float* __restrict__ src, bf16* __restrict__ dst, long long n) {
  for (long long i = (long long)blockIdx.x * 256 + threadIdx.x; i < n;
       i += (long long)gridDim.x * 256)
    dst[i] = (bf16)src[i];
}

__global__ void k_zero_f32(float* __restrict__ p, long long n) {
  for (long long i = (long long)blockIdx.x * 256 + threadIdx.x; i < n;
       i += (long long)gridDim.x * 256)
    p[i] = 0.0f;
}

__global__ __launch_bounds__(256) void k_sumsq(const bf16* __restrict__ x, long long n,
                                               float* __restrict__ out) {
  __shared__ float red[256];
  const int tid = threadIdx.x;
  float s = 0.0f;
  for (long long i = (long long)blockIdx.x * 256 + tid; i < n;
       i += (long long)gridDim.x * 256) {
    float v = (float)x[i];
    s += v * v;
  }
  red[tid] = s; __syncthreads();
  for (int st = 128; st > 0; st >>= 1) { if (tid < st) red[tid] += red[tid + st]; __syncthreads(); }
  if (tid == 0) atomicAdd(out, red[0]);
}

__global__ __launch_bounds__(256) void k_colsum(const bf16* __restrict__ x, int rows, int ld,
                                                float* __restrict__ out) {
  const int col = blockIdx.x * 256 + threadIdx.x;
  float s = 0.0f;
  for (int r = blockIdx.y; r < rows; r += gridDim.y) s += (float)x[(size_t)r * ld + col];
  atomicAdd(&out[col], s);
}

// LN + ReLU row kernel: 512 cols, 256 threads (2 per thread)
__global__ __launch_bounds__(256) void k_ln_relu(
    const float* __restrict__ Y, const float* __restrict__ g, const float* __restrict__ b,
    float* __restrict__ Hout, bf16* __restrict__ Hbf, int Nn)
{
  const int n = blockIdx.x, tid = threadIdx.x;
  __shared__ float red[256];
  __shared__ float sh[2];
  const float* yr = Y + (size_t)n * 512;
  float t0 = yr[tid], t1 = yr[tid + 256];
  red[tid] = t0 + t1; __syncthreads();
  for (int st = 128; st > 0; st >>= 1) { if (tid < st) red[tid] += red[tid + st]; __syncthreads(); }
  if (tid == 0) sh[0] = red[0] * (1.0f / 512.0f);
  __syncthreads();
  float mu = sh[0], e0 = t0 - mu, e1 = t1 - mu;
  red[tid] = e0 * e0 + e1 * e1; __syncthreads();
  for (int st = 128; st > 0; st >>= 1) { if (tid < st) red[tid] += red[tid + st]; __syncthreads(); }
  if (tid == 0) sh[1] = red[0] * (1.0f / 512.0f);
  __syncthreads();
  float rstd = rsqrtf(sh[1] + LN_EPS);
  float o0 = e0 * rstd * g[tid] + b[tid];             if (o0 < 0.0f) o0 = 0.0f;
  float o1 = e1 * rstd * g[tid + 256] + b[tid + 256]; if (o1 < 0.0f) o1 = 0.0f;
  Hout[(size_t)n * 512 + tid] = o0;       Hout[(size_t)n * 512 + tid + 256] = o1;
  Hbf [(size_t)n * 512 + tid] = (bf16)o0; Hbf [(size_t)n * 512 + tid + 256] = (bf16)o1;
}

// Per-node attention combine + residual + LayerNorm.
// scale = 1/(||q||*||k||) applied post-hoc (GEMMs ran on unnormalized q,k).
__global__ __launch_bounds__(256) void k_combine_ln(
    const bf16* __restrict__ Q, const bf16* __restrict__ V,
    const float* __restrict__ NUM, const float* __restrict__ KSUM,
    const float* __restrict__ SUMS, const float* __restrict__ Hin,
    const float* __restrict__ g, const float* __restrict__ b,
    float* __restrict__ Hout, bf16* __restrict__ Hbf, int Nn)
{
  const int n = blockIdx.x, tid = threadIdx.x;
  __shared__ float red[256];
  __shared__ float sh[4];
  const bf16*  qrow = Q + (size_t)n * 1024;
  const bf16*  vrow = V + (size_t)n * 1024;
  const float* numr = NUM + (size_t)n * 1024;

  float d0 = 0.0f, d1 = 0.0f;
  for (int c = tid; c < 512; c += 256) {
    d0 += (float)qrow[c] * KSUM[c];
    d1 += (float)qrow[512 + c] * KSUM[512 + c];
  }
  red[tid] = d0; __syncthreads();
  for (int st = 128; st > 0; st >>= 1) { if (tid < st) red[tid] += red[tid + st]; __syncthreads(); }
  if (tid == 0) sh[0] = red[0];
  __syncthreads();
  red[tid] = d1; __syncthreads();
  for (int st = 128; st > 0; st >>= 1) { if (tid < st) red[tid] += red[tid + st]; __syncthreads(); }
  if (tid == 0) sh[1] = red[0];
  __syncthreads();

  const float scale = rsqrtf(SUMS[0]) * rsqrtf(SUMS[1]);
  const float Nf = (float)Nn;
  const float den0 = sh[0] * scale + Nf;
  const float den1 = sh[1] * scale + Nf;

  const int c0 = tid, c1 = tid + 256;
  float a0 = 0.5f * ((numr[c0] * scale + Nf * (float)vrow[c0]) / den0 +
                     (numr[512 + c0] * scale + Nf * (float)vrow[512 + c0]) / den1);
  float a1 = 0.5f * ((numr[c1] * scale + Nf * (float)vrow[c1]) / den0 +
                     (numr[512 + c1] * scale + Nf * (float)vrow[512 + c1]) / den1);
  float t0 = 0.5f * a0 + 0.5f * Hin[(size_t)n * 512 + c0];
  float t1 = 0.5f * a1 + 0.5f * Hin[(size_t)n * 512 + c1];

  red[tid] = t0 + t1; __syncthreads();
  for (int st = 128; st > 0; st >>= 1) { if (tid < st) red[tid] += red[tid + st]; __syncthreads(); }
  if (tid == 0) sh[2] = red[0] * (1.0f / 512.0f);
  __syncthreads();
  float mu = sh[2], e0 = t0 - mu, e1 = t1 - mu;
  red[tid] = e0 * e0 + e1 * e1; __syncthreads();
  for (int st = 128; st > 0; st >>= 1) { if (tid < st) red[tid] += red[tid + st]; __syncthreads(); }
  if (tid == 0) sh[3] = red[0] * (1.0f / 512.0f);
  __syncthreads();
  float rstd = rsqrtf(sh[3] + LN_EPS);
  float o0 = e0 * rstd * g[c0] + b[c0];
  float o1 = e1 * rstd * g[c1] + b[c1];
  Hout[(size_t)n * 512 + c0] = o0;       Hout[(size_t)n * 512 + c1] = o1;
  Hbf [(size_t)n * 512 + c0] = (bf16)o0; Hbf [(size_t)n * 512 + c1] = (bf16)o1;
}

// ---------------------------------------------------------------------------
extern "C" void kernel_launch(void* const* d_in, const int* in_sizes, int n_in,
                              void* d_out, int out_size, void* d_ws, size_t ws_size,
                              hipStream_t stream)
{
  const float* gnn_x = (const float*)d_in[1];
  const float* fc0_w = (const float*)d_in[4];
  const float* fc0_b = (const float*)d_in[5];
  const float* ln0_g = (const float*)d_in[6];
  const float* ln0_b = (const float*)d_in[7];
  const float* wq_w[2] = {(const float*)d_in[8],  (const float*)d_in[16]};
  const float* wq_b[2] = {(const float*)d_in[9],  (const float*)d_in[17]};
  const float* wk_w[2] = {(const float*)d_in[10], (const float*)d_in[18]};
  const float* wk_b[2] = {(const float*)d_in[11], (const float*)d_in[19]};
  const float* wv_w[2] = {(const float*)d_in[12], (const float*)d_in[20]};
  const float* wv_b[2] = {(const float*)d_in[13], (const float*)d_in[21]};
  const float* ln_g[2] = {(const float*)d_in[14], (const float*)d_in[22]};
  const float* ln_b[2] = {(const float*)d_in[15], (const float*)d_in[23]};
  const float* wr_w = (const float*)d_in[24];
  const float* fc_w = (const float*)d_in[25];
  const float* fc_b = (const float*)d_in[26];
  float* out = (float*)d_out;

  // -------- workspace carving --------
  char* base = (char*)d_ws;
  size_t off = 0;
  auto take = [&](size_t bytes) -> void* {
    void* r = base + off;
    off = (off + bytes + 255) & ~(size_t)255;
    return r;
  };
  float* H32 = (float*)take((size_t)NN * DIN * 4);
  bf16*  HBF = (bf16*) take((size_t)NN * DIN * 2);
  bf16*  Qb  = (bf16*) take((size_t)NN * HC * 2);
  bf16*  Kb  = (bf16*) take((size_t)NN * HC * 2);
  bf16*  Vb  = (bf16*) take((size_t)NN * HC * 2);
  float* NUM = (float*)take((size_t)NN * HC * 4);
  float* T32 = NUM;            // alias: layer-0 pre-LN staging (disjoint lifetime)
  bf16*  TBF = (bf16*)NUM;     // alias: final h@wr in bf16 (disjoint lifetime)
  float* KVS = (float*)take((size_t)(2 * 512 * 512 + 1024 + 64) * 4);
  float* KSUM = KVS + 2 * 512 * 512;
  float* SUMS = KSUM + 1024;
  bf16* KVSBF = (bf16*)take((size_t)2 * 512 * 512 * 2);
  bf16* fc0bf = (bf16*)take((size_t)512 * 512 * 2);
  bf16 *wqbf[2], *wkbf[2], *wvbf[2];
  for (int l = 0; l < 2; ++l) {
    wqbf[l] = (bf16*)take((size_t)512 * HC * 2);
    wkbf[l] = (bf16*)take((size_t)512 * HC * 2);
    wvbf[l] = (bf16*)take((size_t)512 * HC * 2);
  }
  bf16* wrbf = (bf16*)take((size_t)512 * 512 * 2);
  bf16* fcbf = (bf16*)take((size_t)512 * 512 * 2);

  auto cvt = [&](const float* s, bf16* d, long long n) {
    int blocks = (int)((n + 255) / 256); if (blocks > 4096) blocks = 4096;
    k_f32_to_bf16<<<blocks, 256, 0, stream>>>(s, d, n);
  };
  auto gemm = [&](const bf16* A, int lda, const bf16* B, int ldb, const float* bias,
                  float* Cf, bf16* Cb, int ldc, int M, int Ncols, int K) {
    dim3 g(Ncols / 128, (M + 127) / 128);
    k_gemm_bf16<<<g, 256, 0, stream>>>(A, B, bias, Cf, Cb, M, Ncols, K, lda, ldb, ldc);
  };

  // -------- weight + input conversion --------
  cvt(fc0_w, fc0bf, (long long)512 * 512);
  for (int l = 0; l < 2; ++l) {
    cvt(wq_w[l], wqbf[l], (long long)512 * HC);
    cvt(wk_w[l], wkbf[l], (long long)512 * HC);
    cvt(wv_w[l], wvbf[l], (long long)512 * HC);
  }
  cvt(wr_w, wrbf, (long long)512 * 512);
  cvt(fc_w, fcbf, (long long)512 * 512);
  cvt(gnn_x, HBF, (long long)NN * DIN);

  // -------- h = relu(LN(gnn_x @ fc0 + b)) --------
  gemm(HBF, DIN, fc0bf, DIN, fc0_b, T32, nullptr, DIN, NN, DIN, DIN);
  k_ln_relu<<<NN, 256, 0, stream>>>(T32, ln0_g, ln0_b, H32, HBF, NN);

  // -------- two TransConv layers --------
  for (int l = 0; l < 2; ++l) {
    gemm(HBF, DIN, wqbf[l], HC, wq_b[l], nullptr, Qb, HC, NN, HC, DIN);
    gemm(HBF, DIN, wkbf[l], HC, wk_b[l], nullptr, Kb, HC, NN, HC, DIN);
    gemm(HBF, DIN, wvbf[l], HC, wv_b[l], nullptr, Vb, HC, NN, HC, DIN);

    k_zero_f32<<<2048, 256, 0, stream>>>(KVS, (long long)(2 * 512 * 512 + 1024 + 64));
    k_sumsq<<<2048, 256, 0, stream>>>(Qb, (long long)NN * HC, &SUMS[0]);
    k_sumsq<<<2048, 256, 0, stream>>>(Kb, (long long)NN * HC, &SUMS[1]);
    k_colsum<<<dim3(HC / 256, 128), 256, 0, stream>>>(Kb, NN, HC, KSUM);

    k_gemm_atb<<<dim3(4, 4, 2 * ATB_SPLITS), 256, 0, stream>>>(Kb, Vb, KVS, NN, HC);
    cvt(KVS, KVSBF, (long long)2 * 512 * 512);

    for (int h = 0; h < 2; ++h)
      gemm(Qb + h * 512, HC, KVSBF + (size_t)h * 512 * 512, 512, nullptr,
           NUM + h * 512, nullptr, HC, NN, 512, 512);

    k_combine_ln<<<NN, 256, 0, stream>>>(Qb, Vb, NUM, KSUM, SUMS, H32,
                                         ln_g[l], ln_b[l], H32, HBF, NN);
  }

  // -------- out = (h @ wr) @ fc + fc_b --------
  gemm(HBF, DIN, wrbf, DIN, nullptr, nullptr, TBF, DIN, NN, DIN, DIN);
  gemm(TBF, DIN, fcbf, DIN, fc_b, out, nullptr, DIN, NN, DIN, DIN);
}
